// MultiHeadedAttention_52819507806316
// MI455X (gfx1250) — matrix operand order repaired and checked by reading, not compiled, not run on hardware
//
#include <hip/hip_runtime.h>

#define BD 4
#define DM 1024
#define NN 2048
#define HH 16
#define HDIM 64

typedef __attribute__((ext_vector_type(16))) _Float16 v16h;
typedef __attribute__((ext_vector_type(8)))  _Float16 v8h;
typedef __attribute__((ext_vector_type(4)))  _Float16 v4h;
typedef __attribute__((ext_vector_type(8)))  float    v8f;
typedef __attribute__((ext_vector_type(4)))  unsigned int u32x4;
typedef __attribute__((ext_vector_type(8)))  int      i32x8;
typedef __attribute__((ext_vector_type(4)))  int      i32x4;

#if defined(__has_builtin)
#if __has_builtin(__builtin_amdgcn_tensor_load_to_lds) && __has_builtin(__builtin_amdgcn_s_wait_tensorcnt)
#define HAVE_TDM 1
#endif
#endif

// ---- WMMA fragment helpers (CDNA5 16x16x32 f16, wave32) -------------------
static __device__ __forceinline__ v16h ld_a_frag(const _Float16* base, int lda, int lane) {
  const int m = lane & 15, hf = lane >> 4;
  const _Float16* p = base + m * lda + 8 * hf;
  v8h lo = *(const v8h*)(p);
  v8h hi = *(const v8h*)(p + 16);
  v16h r;
#pragma unroll
  for (int i = 0; i < 8; ++i) { r[i] = lo[i]; r[i + 8] = hi[i]; }
  return r;
}
static __device__ __forceinline__ v16h ld_bt_frag(const _Float16* baseT, int ldb, int lane) {
  const int n = lane & 15, hf = lane >> 4;
  const _Float16* p = baseT + n * ldb + 8 * hf;
  v8h lo = *(const v8h*)(p);
  v8h hi = *(const v8h*)(p + 16);
  v16h r;
#pragma unroll
  for (int i = 0; i < 8; ++i) { r[i] = lo[i]; r[i + 8] = hi[i]; }
  return r;
}
static __device__ __forceinline__ v8f wmma_f16(v16h a, v16h b, v8f c) {
  return __builtin_amdgcn_wmma_f32_16x16x32_f16(false, a, false, b, (short)0, c, false, false);
}

#ifdef HAVE_TDM
// 2D tile load via Tensor Data Mover. Descriptor per CDNA5 ISA ch.8 (D#).
// Units are data_size elements (2 bytes). One wave issues; EXEC ignored.
static __device__ __forceinline__ void tdm_load_2d_f16(
    unsigned int lds_off, unsigned long long gaddr,
    unsigned int tensor_d0, unsigned int tensor_d1,
    unsigned int tile_d0, unsigned int tile_d1, unsigned int d0_stride)
{
  u32x4 g0;
  g0[0] = 1u;                                        // count=1, user mode
  g0[1] = lds_off;                                   // LDS byte address
  g0[2] = (unsigned int)gaddr;                       // global addr [31:0]
  g0[3] = (unsigned int)((gaddr >> 32) & 0x1FFFFFFu) // global addr [56:32]
        | (2u << 30);                                // type = 2 ("image")
  i32x8 g1;
  g1[0] = (int)(1u << 16);                           // data_size = 1 -> 2 bytes
  g1[1] = (int)((tensor_d0 & 0xFFFFu) << 16);        // tensor_dim0[15:0]
  g1[2] = (int)((tensor_d0 >> 16) | ((tensor_d1 & 0xFFFFu) << 16));
  g1[3] = (int)((tensor_d1 >> 16) | (tile_d0 << 16));
  g1[4] = (int)tile_d1;                              // tile_dim1 (tile_dim2=0)
  g1[5] = (int)d0_stride;                            // tensor_dim0_stride[31:0]
  g1[6] = 0; g1[7] = 0;
  i32x4 z4 = (i32x4)0;
#if __clang_major__ >= 23
  __builtin_amdgcn_tensor_load_to_lds(g0, g1, z4, z4, (i32x8)0, 0);
#else
  __builtin_amdgcn_tensor_load_to_lds(g0, g1, z4, z4, 0);
#endif
}
#endif

// ---- GEMM (+bias, + RoPE / layout change), C tile 64(o) x 128(n), Kstep 64 -
// MODE 0/1: dst f16 (B,H,N,HD) = rope(W*X+b)      X fp32 (B,D,N)
// MODE 2  : dst f16 (B,H,HD,N) = W*X+b  (V^T)     X fp32 (B,D,N)
// MODE 3  : dst f32 (B,D,N)    = W*X+b            X f16  (B,H,N,HD)
template <int MODE>
__global__ __launch_bounds__(256) void gemm_rope_kernel(
    const float* __restrict__ W, const void* __restrict__ Xv,
    const float* __restrict__ bias, const float* __restrict__ enc,
    void* __restrict__ dstv)
{
  __shared__ alignas(32) _Float16 a_lds[64 * 64];    // W tile (o,k)       8KB
  __shared__ alignas(32) _Float16 bt_lds[128 * 64];  // X^T tile (n,k)    16KB
  __shared__ alignas(16) float    c_lds[64 * 128];   //                   32KB

  const int tid  = threadIdx.x;
  const int lane = tid & 31, wave = tid >> 5;
  const int o0 = blockIdx.y * 64, n0 = blockIdx.x * 128, bz = blockIdx.z;
  const int wm  = wave & 3;
  const int wn0 = (wave >> 2) * 4;

  const int ol_a = tid >> 2, ka = (tid & 3) * 16;    // A staging: 16 contig k
  const int il_b = tid >> 2, nsb = (tid & 3) * 32;   // B staging: 32 contig n

  v8f acc[4];
#pragma unroll
  for (int t = 0; t < 4; ++t)
#pragma unroll
    for (int r = 0; r < 8; ++r) acc[t][r] = 0.f;

  for (int kb = 0; kb < DM; kb += 64) {
    __syncthreads();
    { // A tile: 4x float4 -> 2x ds_store_b128
      const float4* wp = (const float4*)(W + (size_t)(o0 + ol_a) * DM + kb + ka);
#pragma unroll
      for (int s = 0; s < 2; ++s) {
        float4 f0 = wp[2 * s], f1 = wp[2 * s + 1];
        v8h p;
        p[0] = (_Float16)f0.x; p[1] = (_Float16)f0.y; p[2] = (_Float16)f0.z; p[3] = (_Float16)f0.w;
        p[4] = (_Float16)f1.x; p[5] = (_Float16)f1.y; p[6] = (_Float16)f1.z; p[7] = (_Float16)f1.w;
        *(v8h*)(a_lds + ol_a * 64 + ka + 8 * s) = p;
      }
      if (kb + 64 < DM)
        __builtin_prefetch(W + (size_t)(o0 + ol_a) * DM + kb + 64 + ka, 0, 3);
    }
    if constexpr (MODE < 3) {   // B^T tile from fp32 (B,D,N): transpose scatter
      const float* X = (const float*)Xv;
      const float* xp = X + ((size_t)bz * DM + kb + il_b) * NN + n0 + nsb;
#pragma unroll
      for (int s = 0; s < 8; ++s) {
        float4 f = *(const float4*)(xp + s * 4);
        bt_lds[(nsb + s * 4 + 0) * 64 + il_b] = (_Float16)f.x;
        bt_lds[(nsb + s * 4 + 1) * 64 + il_b] = (_Float16)f.y;
        bt_lds[(nsb + s * 4 + 2) * 64 + il_b] = (_Float16)f.z;
        bt_lds[(nsb + s * 4 + 3) * 64 + il_b] = (_Float16)f.w;
      }
      if (kb + 64 < DM) __builtin_prefetch(xp + 64ull * NN, 0, 3);
    } else {                    // gather f16 (B,H,N,HD): 4 adjacent hd per (n,h)
      const _Float16* X16 = (const _Float16*)Xv;
      const int hdb = kb >> 4;
#pragma unroll
      for (int e = 0; e < 8; ++e) {
        int p  = tid + e * 256;            // 2048 (n,h) pairs
        int nl = p & 127, h = p >> 7;
        v4h u = *(const v4h*)(X16 + (((size_t)bz * HH + h) * NN + n0 + nl) * HDIM + hdb);
        bt_lds[nl * 64 +  0 + h] = u[0];
        bt_lds[nl * 64 + 16 + h] = u[1];
        bt_lds[nl * 64 + 32 + h] = u[2];
        bt_lds[nl * 64 + 48 + h] = u[3];
      }
    }
    __syncthreads();
#pragma unroll
    for (int s = 0; s < 2; ++s) {
      v16h a = ld_a_frag(a_lds + wm * 16 * 64 + s * 32, 64, lane);
#pragma unroll
      for (int t = 0; t < 4; ++t) {
        v16h b = ld_bt_frag(bt_lds + (wn0 + t) * 16 * 64 + s * 32, 64, lane);
        acc[t] = wmma_f16(a, b, acc[t]);
      }
    }
  }
  { // park C fragments for the layout-changing epilogue
    const int cl = lane & 15, rh = (lane >> 4) * 8;
#pragma unroll
    for (int t = 0; t < 4; ++t)
#pragma unroll
      for (int r = 0; r < 8; ++r)
        c_lds[(wm * 16 + rh + r) * 128 + (wn0 + t) * 16 + cl] = acc[t][r];
  }
  __syncthreads();

  if constexpr (MODE <= 1) {
    _Float16* dst = (_Float16*)dstv;
    const int hdb = o0 >> 4;
#pragma unroll
    for (int e = 0; e < 8; ++e) {
      int p  = tid + e * 256;              // 2048 (n,h) pairs
      int nl = p & 127, h = p >> 7;
      int n  = n0 + nl;
      float v0 = c_lds[(h     ) * 128 + nl] + bias[o0 + h];
      float v1 = c_lds[(h + 16) * 128 + nl] + bias[o0 + h + 16];
      float v2 = c_lds[(h + 32) * 128 + nl] + bias[o0 + h + 32];
      float v3 = c_lds[(h + 48) * 128 + nl] + bias[o0 + h + 48];
      float4 f0 = *(const float4*)(enc + (size_t)n * HDIM + hdb);
      float4 f1 = *(const float4*)(enc + ((size_t)NN + n) * HDIM + hdb);
      v4h r;
      r[0] = (_Float16)(v0 * f0.x - v1 * f1.x);   // hd even
      r[1] = (_Float16)(v1 * f0.y + v0 * f1.y);   // hd odd
      r[2] = (_Float16)(v2 * f0.z - v3 * f1.z);
      r[3] = (_Float16)(v3 * f0.w + v2 * f1.w);
      *(v4h*)(dst + (((size_t)bz * HH + h) * NN + n) * HDIM + hdb) = r;
    }
  } else if constexpr (MODE == 2) {
    _Float16* dst = (_Float16*)dstv;
#pragma unroll
    for (int e = 0; e < 4; ++e) {
      int q  = tid + e * 256;              // 1024 8-wide units
      int ol = q >> 4, ns = (q & 15) * 8;
      int o  = o0 + ol, hd = o >> 4, h = o & 15;
      float b = bias[o];
      const float* cr = c_lds + ol * 128 + ns;
      v8h r;
#pragma unroll
      for (int i = 0; i < 8; ++i) r[i] = (_Float16)(cr[i] + b);
      *(v8h*)(dst + (((size_t)bz * HH + h) * HDIM + hd) * NN + n0 + ns) = r;
    }
  } else {
    float* dst = (float*)dstv;
#pragma unroll
    for (int e = 0; e < 8; ++e) {
      int q  = tid + e * 256;              // 2048 quads
      int ol = q >> 5, ns = (q & 31) * 4;
      float b = bias[o0 + ol];
      const float* cr = c_lds + ol * 128 + ns;
      float4 res = { cr[0] + b, cr[1] + b, cr[2] + b, cr[3] + b };
      *(float4*)(dst + ((size_t)bz * DM + o0 + ol) * NN + n0 + ns) = res;
    }
  }
}

// ---- Flash attention: q/x f16 (B,H,N,HD), k f16 (B,H,N,HD), v f16 (B,H,HD,N)
// Double-buffered TDM staging; register-resident online softmax via shuffles.
__global__ __launch_bounds__(256) void flash_attn_kernel(
    const _Float16* __restrict__ q, const _Float16* __restrict__ k,
    const _Float16* __restrict__ v, _Float16* __restrict__ x)
{
  __shared__ alignas(32) _Float16 k_lds[2][64 * 64];   // (m, hd)  16KB
  __shared__ alignas(32) _Float16 vt_lds[2][64 * 64];  // (hd, m)  16KB
  __shared__ alignas(32) _Float16 p_lds[8][16 * 64];   // P strips 16KB

  const int tid  = threadIdx.x;
  const int lane = tid & 31, wave = tid >> 5;
  const int n0 = blockIdx.x * 128;
  const int h  = blockIdx.y;
  const int bz = blockIdx.z;

  const size_t headoff = ((size_t)bz * HH + h) * (size_t)NN * HDIM;
  const _Float16* qh  = q + headoff;
  const _Float16* kh  = k + headoff;
  const _Float16* vth = v + headoff;    // (HD, N) rows for this head

  // Q fragments in registers, pre-scaled by 1/sqrt(HD) = 0.125
  const int am = lane & 15, ah = lane >> 4;
  const _Float16* qrow = qh + (size_t)(n0 + wave * 16 + am) * HDIM;
  v16h aq[2];
#pragma unroll
  for (int s = 0; s < 2; ++s) {
    v8h lo = *(const v8h*)(qrow + s * 32 + 8 * ah);
    v8h hi = *(const v8h*)(qrow + s * 32 + 16 + 8 * ah);
#pragma unroll
    for (int i = 0; i < 8; ++i) {
      aq[s][i]     = (_Float16)(0.125f * (float)lo[i]);
      aq[s][i + 8] = (_Float16)(0.125f * (float)hi[i]);
    }
  }

  v8f o_acc[4];
#pragma unroll
  for (int t = 0; t < 4; ++t)
#pragma unroll
    for (int r = 0; r < 8; ++r) o_acc[t][r] = 0.f;

  float mrow[8], lrow[8];
#pragma unroll
  for (int r = 0; r < 8; ++r) { mrow[r] = -3.0e38f; lrow[r] = 0.f; }

  _Float16* p_strip = p_lds[wave];
  const int cl = lane & 15, rh = (lane >> 4) * 8;

#ifdef HAVE_TDM
  unsigned int klds_off[2], vlds_off[2];
#pragma unroll
  for (int b = 0; b < 2; ++b) {
    klds_off[b] = (unsigned int)(unsigned long long)(void*)&k_lds[b][0];
    vlds_off[b] = (unsigned int)(unsigned long long)(void*)&vt_lds[b][0];
  }
  if (wave == 0) {  // prologue: tile 0 into buffer 0
    tdm_load_2d_f16(klds_off[0], (unsigned long long)(const void*)kh,
                    HDIM, NN, 64, 64, HDIM);
    tdm_load_2d_f16(vlds_off[0], (unsigned long long)(const void*)vth,
                    NN, HDIM, 64, 64, NN);
  }
#else
  const int sr = tid >> 2, scs = (tid & 3) * 16;
#endif

  for (int j = 0; j < NN / 64; ++j) {
    const int cur = j & 1;
    __syncthreads();                       // buf[1-cur] consumers (tile j-1) done
#ifdef HAVE_TDM
    if (wave == 0) {
      if (j + 1 < NN / 64) {               // DMA tile j+1 under tile j's compute
        tdm_load_2d_f16(klds_off[1 - cur],
            (unsigned long long)(const void*)(kh + (size_t)(j + 1) * 64 * HDIM),
            HDIM, NN, 64, 64, HDIM);
        tdm_load_2d_f16(vlds_off[1 - cur],
            (unsigned long long)(const void*)(vth + (size_t)(j + 1) * 64),
            NN, HDIM, 64, 64, NN);
        __builtin_amdgcn_s_wait_tensorcnt(2);  // in-order: tile j's 2 ops done
      } else {
        __builtin_amdgcn_s_wait_tensorcnt(0);
      }
    }
#else
    {
      const _Float16* kp = kh + (size_t)(j * 64 + sr) * HDIM + scs;
      *(v8h*)(&k_lds[cur][0] + sr * 64 + scs)     = *(const v8h*)kp;
      *(v8h*)(&k_lds[cur][0] + sr * 64 + scs + 8) = *(const v8h*)(kp + 8);
      const _Float16* vp = vth + (size_t)sr * NN + j * 64 + scs;
      *(v8h*)(&vt_lds[cur][0] + sr * 64 + scs)     = *(const v8h*)vp;
      *(v8h*)(&vt_lds[cur][0] + sr * 64 + scs + 8) = *(const v8h*)(vp + 8);
      if (j + 1 < NN / 64) {
        __builtin_prefetch(kh + (size_t)(j + 1) * 64 * HDIM + tid * 16, 0, 3);
        __builtin_prefetch(vth + (size_t)sr * NN + (j + 1) * 64, 0, 3);
      }
    }
#endif
    __syncthreads();                       // tile j visible to all waves

    // ---- S = (Q/8) K^T (16 x 64 strip per wave), f32 accum in fragments
    v8f s_acc[4];
#pragma unroll
    for (int t = 0; t < 4; ++t)
#pragma unroll
      for (int r = 0; r < 8; ++r) s_acc[t][r] = 0.f;
#pragma unroll
    for (int s = 0; s < 2; ++s) {
#pragma unroll
      for (int t = 0; t < 4; ++t) {
        v16h bfr = ld_bt_frag(&k_lds[cur][0] + t * 16 * 64 + s * 32, 64, lane);
        s_acc[t] = wmma_f16(aq[s], bfr, s_acc[t]);
      }
    }

    // ---- register softmax: rows live in 16-lane halves; xor-shuffle reduce
    float al[8];
#pragma unroll
    for (int r = 0; r < 8; ++r) {
      float mx = fmaxf(fmaxf(s_acc[0][r], s_acc[1][r]),
                       fmaxf(s_acc[2][r], s_acc[3][r]));
      mx = fmaxf(mx, __shfl_xor(mx, 1, 32));
      mx = fmaxf(mx, __shfl_xor(mx, 2, 32));
      mx = fmaxf(mx, __shfl_xor(mx, 4, 32));
      mx = fmaxf(mx, __shfl_xor(mx, 8, 32));
      float nm = fmaxf(mrow[r], mx);
      al[r] = __expf(mrow[r] - nm);
      mrow[r] = nm;
      float p0 = __expf(s_acc[0][r] - nm);
      float p1 = __expf(s_acc[1][r] - nm);
      float p2 = __expf(s_acc[2][r] - nm);
      float p3 = __expf(s_acc[3][r] - nm);
      p_strip[(rh + r) * 64 +  0 + cl] = (_Float16)p0;
      p_strip[(rh + r) * 64 + 16 + cl] = (_Float16)p1;
      p_strip[(rh + r) * 64 + 32 + cl] = (_Float16)p2;
      p_strip[(rh + r) * 64 + 48 + cl] = (_Float16)p3;
      float sm = (p0 + p1) + (p2 + p3);
      sm += __shfl_xor(sm, 1, 32);
      sm += __shfl_xor(sm, 2, 32);
      sm += __shfl_xor(sm, 4, 32);
      sm += __shfl_xor(sm, 8, 32);
      lrow[r] = lrow[r] * al[r] + sm;
    }

    // ---- O = O*alpha + P*V  (P strip is wave-private: no block barrier needed)
#pragma unroll
    for (int t = 0; t < 4; ++t)
#pragma unroll
      for (int r = 0; r < 8; ++r)
        o_acc[t][r] *= al[r];
#pragma unroll
    for (int s = 0; s < 2; ++s) {
      v16h ap = ld_a_frag(p_strip + s * 32, 64, lane);
#pragma unroll
      for (int t = 0; t < 4; ++t) {
        v16h bfr = ld_bt_frag(&vt_lds[cur][0] + t * 16 * 64 + s * 32, 64, lane);
        o_acc[t] = wmma_f16(ap, bfr, o_acc[t]);
      }
    }
  }

  // ---- normalize (1/l in registers) and store f16 (B,H,N,HD)
#pragma unroll
  for (int r = 0; r < 8; ++r) {
    float linv = 1.0f / lrow[r];
    int row = n0 + wave * 16 + rh + r;
#pragma unroll
    for (int t = 0; t < 4; ++t)
      x[headoff + (size_t)row * HDIM + t * 16 + cl] =
          (_Float16)(o_acc[t][r] * linv);
  }
}

extern "C" void kernel_launch(void* const* d_in, const int* in_sizes, int n_in,
                              void* d_out, int out_size, void* d_ws, size_t ws_size,
                              hipStream_t stream) {
  (void)in_sizes; (void)n_in; (void)out_size; (void)ws_size;
  const float* query = (const float*)d_in[0];
  const float* key   = (const float*)d_in[1];
  const float* value = (const float*)d_in[2];
  const float* enc   = (const float*)d_in[3];
  const float* Wq = (const float*)d_in[4];
  const float* bq = (const float*)d_in[5];
  const float* Wk = (const float*)d_in[6];
  const float* bk = (const float*)d_in[7];
  const float* Wv = (const float*)d_in[8];
  const float* bv = (const float*)d_in[9];
  const float* Wm = (const float*)d_in[10];
  const float* bm = (const float*)d_in[11];

  const size_t per = (size_t)BD * HH * NN * HDIM;   // 8M f16 elems per buffer
  _Float16* qr = (_Float16*)d_ws;
  _Float16* kr = qr + per;
  _Float16* vr = kr + per;      // (B,H,HD,N)
  _Float16* xr = vr + per;

  dim3 blk(256);
  dim3 gg(NN / 128, DM / 64, BD);
  gemm_rope_kernel<0><<<gg, blk, 0, stream>>>(Wq, query, bq, enc, qr);
  gemm_rope_kernel<1><<<gg, blk, 0, stream>>>(Wk, key,   bk, enc, kr);
  gemm_rope_kernel<2><<<gg, blk, 0, stream>>>(Wv, value, bv, enc, vr);
  flash_attn_kernel<<<dim3(NN / 128, HH, BD), blk, 0, stream>>>(qr, kr, vr, xr);
  gemm_rope_kernel<3><<<gg, blk, 0, stream>>>(Wm, xr, bm, enc, d_out);
}
